// MS_Attention_linear_29403346108740
// MI455X (gfx1250) — matrix operand order repaired and checked by reading, not compiled, not run on hardware
//
#include <hip/hip_runtime.h>
#include <hip/hip_bf16.h>

// MS-Attention (linear, no softmax) fused pipeline for gfx1250 (MI455X).
// - Reassociates (q@k^T)@v -> q@(k^T@v): attention drops 8.6 GF -> 0.27 GF.
// - Spiked activations are exact multiples of 1/4 in [0,1] => exact in FP16,
//   so all GEMMs use v_wmma_f32_16x16x32_f16 (full-rate CDNA5 XDL path).
// - Conv GEMMs use a 32x64 macro-tile per wave: 8 WMMA per 12 b128 loads,
//   ~2.7x better flop/byte than a 16x16 tile, cutting L2 traffic ~2.7x.

typedef _Float16 f16;
typedef __attribute__((ext_vector_type(16))) _Float16 v16h;
typedef __attribute__((ext_vector_type(8)))  _Float16 v8h;
typedef __attribute__((ext_vector_type(8)))  float    v8f;

#define BB     8
#define CCH    256
#define NN     1024
#define HEADS  8
#define DH     32
#define BN_EPS 1e-5f

__device__ __forceinline__ float spikef(float x) {
    // eval-mode MultiSpike_norm4: round(clamp(x,0,4))/4  (round-half-even)
    x = fminf(fmaxf(x, 0.0f), 4.0f);
    return rintf(x) * 0.25f;
}

__device__ __forceinline__ v8f wmma_f16(v16h a, v16h b, v8f c) {
    return __builtin_amdgcn_wmma_f32_16x16x32_f16(false, a, false, b,
                                                  (short)0, c, false, false);
}

// A 16x32 f16 lane map: lanes 0-15 row M=lane, K{0..15}; lanes 16-31 K{16..31}
__device__ __forceinline__ v16h load_a(const f16* __restrict__ p, int k0, int hl) {
    v8h a0 = *(const v8h*)(p + k0 + hl * 16);
    v8h a1 = *(const v8h*)(p + k0 + hl * 16 + 8);
    return __builtin_shufflevector(a0, a1, 0,1,2,3,4,5,6,7,8,9,10,11,12,13,14,15);
}
// B 32x16 f16 lane map: lanes 0-15 col N=lane, K{0..15}; lanes 16-31 K{16..31}
__device__ __forceinline__ v16h load_b(const f16* __restrict__ p, int k0, int hl) {
    v8h b0 = *(const v8h*)(p + k0 + hl * 16);
    v8h b1 = *(const v8h*)(p + k0 + hl * 16 + 8);
    return __builtin_shufflevector(b0, b1, 0,1,2,3,4,5,6,7,8,9,10,11,12,13,14,15);
}

// ---- Stage 0a: fold BN scale into conv weights (f16) + bias (f32) ----------
__global__ void prep_weights(const float* __restrict__ w, const float* __restrict__ g,
                             const float* __restrict__ be, const float* __restrict__ mn,
                             const float* __restrict__ vr,
                             f16* __restrict__ wt, float* __restrict__ bias) {
    int idx = blockIdx.x * blockDim.x + threadIdx.x;   // idx = o*C + c
    int o = idx >> 8;
    int c = idx & 255;
    float s = g[o] * rsqrtf(vr[o] + BN_EPS);
    wt[idx] = (f16)(w[idx] * s);
    if (c == 0) bias[o] = be[o] - mn[o] * s;
}

// ---- Stage 0b: spike(x) + transpose [B,C,N] -> [B,N,C] f16 -----------------
__global__ void spike_x_kernel(const float* __restrict__ x, f16* __restrict__ xs) {
    int idx = blockIdx.x * blockDim.x + threadIdx.x;   // b*C*N + c*N + n
    int b = idx / (CCH * NN);
    int r = idx - b * (CCH * NN);
    int c = r >> 10;
    int n = r & 1023;
    xs[(size_t)(b * NN + n) * CCH + c] = (f16)spikef(x[idx]);
}

// ---- Stages 1 & 4: [8192 x 256] x [256 x 256] GEMM, optional spike ---------
// 32x64 macro-tile per wave: 2 A fragments x 4 B fragments -> 8 WMMA per
// k-step, K=256 in 8 k-steps (64 WMMA per wave).
__global__ void gemm_conv(const f16* __restrict__ A, const f16* __restrict__ Wt,
                          const float* __restrict__ bias,
                          f16* __restrict__ outh, float* __restrict__ outf,
                          int do_spike) {
    const int lane = threadIdx.x & 31;
    const int hl   = lane >> 4;
    const int lr   = lane & 15;
    const int w    = blockIdx.x * (blockDim.x >> 5) + (threadIdx.x >> 5);
    const int nt   = w & 3;           // 4 col groups of 64 (C = 256)
    const int mt   = w >> 2;          // 256 row groups of 32 (B*N = 8192)
    const int m0   = mt * 32;
    const int n0   = nt * 64;

    const f16* ar0 = A  + (size_t)(m0 + lr) * CCH;
    const f16* ar1 = A  + (size_t)(m0 + 16 + lr) * CCH;
    const f16* br0 = Wt + (size_t)(n0 + lr) * CCH;
    const f16* br1 = Wt + (size_t)(n0 + 16 + lr) * CCH;
    const f16* br2 = Wt + (size_t)(n0 + 32 + lr) * CCH;
    const f16* br3 = Wt + (size_t)(n0 + 48 + lr) * CCH;

    v8f acc[2][4] = {};
    for (int k0 = 0; k0 < CCH; k0 += 32) {
        __builtin_prefetch(ar0 + k0 + 32);              // global_prefetch_b8
        v16h A0 = load_a(ar0, k0, hl);
        v16h A1 = load_a(ar1, k0, hl);
        v16h B0 = load_b(br0, k0, hl);
        v16h B1 = load_b(br1, k0, hl);
        v16h B2 = load_b(br2, k0, hl);
        v16h B3 = load_b(br3, k0, hl);
        acc[0][0] = wmma_f16(A0, B0, acc[0][0]);
        acc[0][1] = wmma_f16(A0, B1, acc[0][1]);
        acc[0][2] = wmma_f16(A0, B2, acc[0][2]);
        acc[0][3] = wmma_f16(A0, B3, acc[0][3]);
        acc[1][0] = wmma_f16(A1, B0, acc[1][0]);
        acc[1][1] = wmma_f16(A1, B1, acc[1][1]);
        acc[1][2] = wmma_f16(A1, B2, acc[1][2]);
        acc[1][3] = wmma_f16(A1, B3, acc[1][3]);
    }

#pragma unroll
    for (int I = 0; I < 2; ++I) {
#pragma unroll
        for (int j = 0; j < 4; ++j) {
            const int col = n0 + j * 16 + lr;           // D col = lane%16
            const float bv = bias[col];
#pragma unroll
            for (int r = 0; r < 8; ++r) {
                const int grow = m0 + I * 16 + r + hl * 8;  // D row = vgpr + 8*half
                float y = acc[I][j][r] + bv;
                if (do_spike) {
                    outh[(size_t)grow * CCH + col] = (f16)spikef(y);
                } else {
                    int b = grow >> 10, n = grow & 1023;
                    outf[((size_t)(b * CCH) + col) * NN + n] = y;   // [B,C,N]
                }
            }
        }
    }
}

// ---- Stage 2: KV[e,d] = sum_n k[n,e]*v[n,d], scaled by 2/sqrt(dh) ----------
// One wave per (b,h); 2x2 tiles of 16x16, K = N = 1024 in 32 WMMA steps.
__global__ void kv_kernel(const f16* __restrict__ K, const f16* __restrict__ V,
                          f16* __restrict__ KVh) {
    const int bh = blockIdx.x;
    const int b = bh >> 3, h = bh & 7;
    const int lane = threadIdx.x & 31;
    const int hl = lane >> 4, lr = lane & 15;
    const f16* kb = K + (size_t)b * NN * CCH + h * DH;
    const f16* vb = V + (size_t)b * NN * CCH + h * DH;

    v8f a00 = {}, a01 = {}, a10 = {}, a11 = {};
    for (int m0 = 0; m0 < NN; m0 += 32) {
        v16h A0, A1, B0, B1;
#pragma unroll
        for (int i = 0; i < 8; ++i) {           // A = K^T (lane-contiguous gather)
            int mlo = m0 + hl * 16 + i;
            int mhi = m0 + hl * 16 + 8 + i;
            A0[i]     = kb[(size_t)mlo * CCH + lr];
            A0[8 + i] = kb[(size_t)mhi * CCH + lr];
            A1[i]     = kb[(size_t)mlo * CCH + 16 + lr];
            A1[8 + i] = kb[(size_t)mhi * CCH + 16 + lr];
        }
#pragma unroll
        for (int i = 0; i < 16; ++i) {          // B = V (lane-contiguous gather)
            int m = m0 + hl * 16 + i;
            B0[i] = vb[(size_t)m * CCH + lr];
            B1[i] = vb[(size_t)m * CCH + 16 + lr];
        }
        a00 = wmma_f16(A0, B0, a00);
        a01 = wmma_f16(A0, B1, a01);
        a10 = wmma_f16(A1, B0, a10);
        a11 = wmma_f16(A1, B1, a11);
    }

    const float sc = 2.0f * 0.1767766952966369f;   // 2 * dh^-0.5
    f16* out = KVh + (size_t)bh * (DH * DH);       // [e][d] row-major
#pragma unroll
    for (int r = 0; r < 8; ++r) {
        int e0 = r + hl * 8;
        out[(size_t)e0 * DH + lr]             = (f16)(a00[r] * sc);
        out[(size_t)e0 * DH + 16 + lr]        = (f16)(a01[r] * sc);
        out[(size_t)(16 + e0) * DH + lr]      = (f16)(a10[r] * sc);
        out[(size_t)(16 + e0) * DH + 16 + lr] = (f16)(a11[r] * sc);
    }
}

// ---- Stage 3: O = spike(q @ KV)  (K = dh = 32 -> single WMMA per tile) -----
__global__ void o_kernel(const f16* __restrict__ Q, const f16* __restrict__ KVh,
                         f16* __restrict__ O) {
    const int bh = blockIdx.x;
    const int b = bh >> 3, h = bh & 7;
    const int lane = threadIdx.x & 31;
    const int hl = lane >> 4, lr = lane & 15;
    const int wv = threadIdx.x >> 5;               // 8 waves per block
    const f16* kv = KVh + (size_t)bh * (DH * DH);
    const f16* qb = Q + (size_t)b * NN * CCH + h * DH;
    f16* ob       = O + (size_t)b * NN * CCH + h * DH;

    v16h B0, B1;                                   // loaded once per block
#pragma unroll
    for (int i = 0; i < 16; ++i) {
        int e = hl * 16 + i;
        B0[i] = kv[e * DH + lr];
        B1[i] = kv[e * DH + 16 + lr];
    }

    for (int t = 0; t < 8; ++t) {
        const int n0 = (wv * 8 + t) * 16;
        const f16* arow = qb + (size_t)(n0 + lr) * CCH;
        v8h a0 = *(const v8h*)(arow + hl * 16);
        v8h a1 = *(const v8h*)(arow + hl * 16 + 8);
        v16h Am = __builtin_shufflevector(a0, a1, 0,1,2,3,4,5,6,7,8,9,10,11,12,13,14,15);
        v8f z = {};
        v8f c0 = wmma_f16(Am, B0, z);
        v8f c1 = wmma_f16(Am, B1, z);
#pragma unroll
        for (int r = 0; r < 8; ++r) {
            int n = n0 + r + hl * 8;
            ob[(size_t)n * CCH + lr]      = (f16)spikef(c0[r]);
            ob[(size_t)n * CCH + 16 + lr] = (f16)spikef(c1[r]);
        }
    }
}

extern "C" void kernel_launch(void* const* d_in, const int* in_sizes, int n_in,
                              void* d_out, int out_size, void* d_ws, size_t ws_size,
                              hipStream_t stream) {
    (void)in_sizes; (void)n_in; (void)out_size; (void)ws_size;
    const float* x = (const float*)d_in[0];

    char* ws = (char*)d_ws;
    size_t off = 0;
    auto take = [&](size_t bytes) -> char* {
        char* p = ws + off;
        off += (bytes + 255) & ~(size_t)255;
        return p;
    };
    const size_t actBytes = (size_t)BB * NN * CCH * sizeof(f16);   // 4 MiB each
    f16*   Xs   = (f16*)take(actBytes);
    f16*   Qb   = (f16*)take(actBytes);
    f16*   Kb   = (f16*)take(actBytes);
    f16*   Vb   = (f16*)take(actBytes);
    f16*   Ob   = (f16*)take(actBytes);
    f16*   Wt   = (f16*)take((size_t)4 * CCH * CCH * sizeof(f16));
    float* bias = (float*)take((size_t)4 * CCH * sizeof(float));
    f16*   KVh  = (f16*)take((size_t)BB * HEADS * DH * DH * sizeof(f16));

    // fold BN into the 4 conv weights (inputs: x, then 5 tensors per conv q,k,v,p)
    for (int i = 0; i < 4; ++i) {
        prep_weights<<<(CCH * CCH) / 256, 256, 0, stream>>>(
            (const float*)d_in[1 + 5 * i], (const float*)d_in[2 + 5 * i],
            (const float*)d_in[3 + 5 * i], (const float*)d_in[4 + 5 * i],
            (const float*)d_in[5 + 5 * i],
            Wt + (size_t)i * CCH * CCH, bias + (size_t)i * CCH);
    }
    spike_x_kernel<<<(BB * CCH * NN) / 256, 256, 0, stream>>>(x, Xs);

    // q/k/v = spike(conv_bn(xs)); 1024 wave-tiles of 32x64 each
    gemm_conv<<<128, 256, 0, stream>>>(Xs, Wt,                     bias,           Qb, nullptr, 1);
    gemm_conv<<<128, 256, 0, stream>>>(Xs, Wt + (size_t)CCH * CCH, bias + CCH,     Kb, nullptr, 1);
    gemm_conv<<<128, 256, 0, stream>>>(Xs, Wt + (size_t)2*CCH*CCH, bias + 2 * CCH, Vb, nullptr, 1);

    // linear attention, reassociated
    kv_kernel<<<BB * HEADS, 32, 0, stream>>>(Kb, Vb, KVh);
    o_kernel<<<BB * HEADS, 256, 0, stream>>>(Qb, KVh, Ob);

    // final conv_bn -> fp32 [B,C,H,W]
    gemm_conv<<<128, 256, 0, stream>>>(Ob, Wt + (size_t)3*CCH*CCH, bias + 3 * CCH,
                                       nullptr, (float*)d_out, 0);
}
